// AttnBlock3DFix_10866267259277
// MI455X (gfx1250) — compile-verified, matrix operand order checked by reading, not run on hardware
//
#include <hip/hip_runtime.h>

// ---------------------------------------------------------------------------
// AttnBlock3D for MI455X (gfx1250): GroupNorm + QKV proj + attention + proj
// x: (1, 512, 8, 64, 64) f32.  C=512, T=8, HW=4096, M=T*HW=32768, GROUPS=32.
// All GEMMs via v_wmma_f32_16x16x32_bf16 (wave32, f32 accum).  Shared tiles
// (K for the stats pass, Q for the PV pass) are staged through LDS with
// global_load_async_to_lds_b128 (ASYNCcnt) overlapping the matrix pipe.
// ---------------------------------------------------------------------------

typedef __attribute__((ext_vector_type(16))) __bf16 v16bf;
typedef __attribute__((ext_vector_type(8)))  __bf16 v8bf;
typedef __attribute__((ext_vector_type(8)))  float  v8f;

#define C_DIM   512
#define T_DIM   8
#define HW_DIM  4096
#define M_DIM   (T_DIM * HW_DIM)   // 32768
#define N_GRP   32
#define ATT_SCALE 0.04419417382415922f   // 512^-0.5 (folded into q projection)

// --- A/B tile load, CDNA5 16-bit 16x32 A-matrix VGPR layout (ISA 7.12.2) ---
__device__ __forceinline__ v16bf load_tile16x32(const __bf16* base, int ld, int lane) {
    int m  = lane & 15;
    int k0 = (lane >> 4) << 3;
    const __bf16* p = base + (size_t)m * ld + k0;
    union { v16bf v; v8bf h[2]; } u;
    u.h[0] = *(const v8bf*)(p);
    u.h[1] = *(const v8bf*)(p + 16);
    return u.v;
}

__device__ __forceinline__ v8f wmma_bf16(v16bf a, v16bf b, v8f c) {
    return __builtin_amdgcn_wmma_f32_16x16x32_bf16(false, a, false, b,
                                                   (short)0, c, false, false);
}

// Async bulk copy: one lane moves 16 bytes global -> LDS (ASYNCcnt-tracked).
__device__ __forceinline__ void async_copy_b128(unsigned lds_off, const void* gaddr) {
    asm volatile("global_load_async_to_lds_b128 %0, %1, off"
                 :: "v"(lds_off), "v"(gaddr) : "memory");
}
__device__ __forceinline__ void wait_async0() {
    asm volatile("s_wait_asynccnt 0x0" ::: "memory");
}

// ---------------------------------------------------------------------------
// GroupNorm statistics, stage 1: 16 partial (sum, sumsq) per group.
// ---------------------------------------------------------------------------
__global__ __launch_bounds__(256)
void gn_partial_kernel(const float* __restrict__ x,
                       float* __restrict__ psum, float* __restrict__ psum2) {
    const int g = blockIdx.x, s = blockIdx.y;
    const float* p = x + (size_t)g * 524288 + (size_t)s * 32768;
    float acc = 0.f, acc2 = 0.f;
    for (int i = threadIdx.x; i < 32768; i += 256) {
        float v = p[i];
        acc += v; acc2 += v * v;
    }
    __shared__ float sb[256], sb2[256];
    sb[threadIdx.x] = acc; sb2[threadIdx.x] = acc2;
    __syncthreads();
    for (int off = 128; off > 0; off >>= 1) {
        if (threadIdx.x < off) {
            sb[threadIdx.x]  += sb[threadIdx.x + off];
            sb2[threadIdx.x] += sb2[threadIdx.x + off];
        }
        __syncthreads();
    }
    if (threadIdx.x == 0) {
        psum[g * 16 + s]  = sb[0];
        psum2[g * 16 + s] = sb2[0];
    }
}

__global__ void gn_finalize_kernel(const float* __restrict__ psum,
                                   const float* __restrict__ psum2,
                                   float* __restrict__ mean, float* __restrict__ rstd) {
    int g = threadIdx.x;
    float s = 0.f, ss = 0.f;
    for (int i = 0; i < 16; ++i) { s += psum[g * 16 + i]; ss += psum2[g * 16 + i]; }
    const float inv = 1.0f / 524288.0f;
    float m = s * inv;
    float var = ss * inv - m * m;
    mean[g] = m;
    rstd[g] = rsqrtf(var + 1e-6f);
}

// ---------------------------------------------------------------------------
// Apply GroupNorm + transpose (c,t,p) -> (t,p,c), f32 -> bf16 (LDS tiled).
// ---------------------------------------------------------------------------
__global__ __launch_bounds__(256)
void gn_apply_kernel(const float* __restrict__ x, const float* __restrict__ gamma,
                     const float* __restrict__ beta, const float* __restrict__ mean,
                     const float* __restrict__ rstd, __bf16* __restrict__ hn) {
    __shared__ float tile[32][33];
    const int t  = blockIdx.z;
    const int p0 = blockIdx.x * 32;
    const int c0 = blockIdx.y * 32;
    #pragma unroll
    for (int i = 0; i < 4; ++i) {
        int c = c0 + threadIdx.y + i * 8;
        tile[threadIdx.y + i * 8][threadIdx.x] =
            x[((size_t)c * T_DIM + t) * HW_DIM + p0 + threadIdx.x];
    }
    __syncthreads();
    const int c = c0 + threadIdx.x;
    const int g = c >> 4;
    const float mu = mean[g], rs = rstd[g];
    const float ga = gamma[c], be = beta[c];
    #pragma unroll
    for (int i = 0; i < 4; ++i) {
        int p = p0 + threadIdx.y + i * 8;
        float v = tile[threadIdx.x][threadIdx.y + i * 8];
        hn[((size_t)t * HW_DIM + p) * C_DIM + c] = (__bf16)((v - mu) * rs * ga + be);
    }
}

// ---------------------------------------------------------------------------
// Weight prep: wT[n][k] = (bf16) w[k][n] for all four 512x512 weights.
// ---------------------------------------------------------------------------
__global__ __launch_bounds__(256)
void prep_w_kernel(const float* __restrict__ wq, const float* __restrict__ wk,
                   const float* __restrict__ wv, const float* __restrict__ wp,
                   __bf16* __restrict__ wqT, __bf16* __restrict__ wkT,
                   __bf16* __restrict__ wvT, __bf16* __restrict__ wpT) {
    int idx = blockIdx.x * 256 + threadIdx.x;
    int n = idx & 511, kk = idx >> 9;
    size_t src = (size_t)kk * C_DIM + n;
    size_t dst = (size_t)n * C_DIM + kk;
    wqT[dst] = (__bf16)wq[src];
    wkT[dst] = (__bf16)wk[src];
    wvT[dst] = (__bf16)wv[src];
    wpT[dst] = (__bf16)wp[src];
}

// ---------------------------------------------------------------------------
// WMMA GEMM, statically software-pipelined (register double-buffered tiles).
// One wave = 16x64 tile (4 N-tiles), 8 waves/block.  grid (N/64, M/128).
// MODE 0: bf16 row-major (t,hw,c)      | MODE 1: bf16 transposed (t,c,hw)
// MODE 2: f32 out = x + bias + val     | MODE 3: bf16 row-major, *ATT_SCALE
// ---------------------------------------------------------------------------
template <int MODE>
__global__ __launch_bounds__(256)
void gemm_kernel(const __bf16* __restrict__ A, const __bf16* __restrict__ WT,
                 const float* __restrict__ bias, __bf16* __restrict__ outb,
                 float* __restrict__ outf, const float* __restrict__ xres) {
    const int lane  = threadIdx.x & 31;
    const int wave  = threadIdx.x >> 5;
    const int mbase = blockIdx.y * 128 + wave * 16;
    const int nbase = blockIdx.x * 64;
    const __bf16* Ap = A + (size_t)mbase * C_DIM;

    v8f acc[4] = {};
    v16bf a = load_tile16x32(Ap, C_DIM, lane);
    v16bf b[4];
    #pragma unroll
    for (int nt = 0; nt < 4; ++nt)
        b[nt] = load_tile16x32(WT + (size_t)(nbase + nt * 16) * C_DIM, C_DIM, lane);

    #pragma unroll
    for (int kc = 0; kc < C_DIM; kc += 32) {
        v16bf an = a;
        v16bf bn[4] = {b[0], b[1], b[2], b[3]};
        if (kc + 32 < C_DIM) {                       // compile-time guard (unrolled)
            an = load_tile16x32(Ap + kc + 32, C_DIM, lane);
            #pragma unroll
            for (int nt = 0; nt < 4; ++nt)
                bn[nt] = load_tile16x32(WT + (size_t)(nbase + nt * 16) * C_DIM + kc + 32,
                                        C_DIM, lane);
        }
        #pragma unroll
        for (int nt = 0; nt < 4; ++nt)
            acc[nt] = wmma_bf16(a, b[nt], acc[nt]);  // independent accum chains
        a = an;
        #pragma unroll
        for (int nt = 0; nt < 4; ++nt) b[nt] = bn[nt];
    }

    const int hi = lane >> 4;
    const int nn = lane & 15;
    #pragma unroll
    for (int nt = 0; nt < 4; ++nt) {
        int n = nbase + nt * 16 + nn;
        float bv = bias[n];
        #pragma unroll
        for (int r = 0; r < 8; ++r) {
            int m = mbase + r + 8 * hi;
            float val = acc[nt][r] + bv;
            if (MODE == 0) {
                outb[(size_t)m * C_DIM + n] = (__bf16)val;
            } else if (MODE == 3) {
                outb[(size_t)m * C_DIM + n] = (__bf16)(val * ATT_SCALE);
            } else if (MODE == 1) {
                int t = m >> 12, p = m & 4095;
                outb[((size_t)t * C_DIM + n) * HW_DIM + p] = (__bf16)val;
            } else {
                int t = m >> 12, p = m & 4095;
                size_t idx = ((size_t)n * T_DIM + t) * HW_DIM + p;
                outf[idx] = xres[idx] + val;
            }
        }
    }
}

// ---------------------------------------------------------------------------
// Attention pass 1: running row-max / row-sumexp, 4 query blocks per block.
// K tiles (16 keys x 512 ch = contiguous 16 KB) are shared by all 4 waves and
// staged via ping-pong global_load_async_to_lds_b128 (buffer 0 = even 16-key
// tile, buffer 1 = odd).  Two S tiles per 32-key round share ONE combined
// shfl reduction (halves softmax VALU).  grid (HW/64, T), block 128.
// ---------------------------------------------------------------------------
__global__ __launch_bounds__(128)
void attn_stats_kernel(const __bf16* __restrict__ q, const __bf16* __restrict__ k,
                       float* __restrict__ mstat, float* __restrict__ lstat) {
    __shared__ __bf16 sK[2][16 * C_DIM];             // 2 x 16 KB
    const int tid   = threadIdx.x;
    const int lane  = tid & 31;
    const int wave  = tid >> 5;
    const int t     = blockIdx.y;
    const int qbase = (blockIdx.x * 4 + wave) * 16;
    const __bf16* qp = q + ((size_t)t * HW_DIM + qbase) * C_DIM;
    const __bf16* kp = k + (size_t)t * HW_DIM * C_DIM;
    const unsigned sK_lds = (unsigned)(uintptr_t)&sK[0][0];

    v16bf aq[16];
    #pragma unroll
    for (int kc = 0; kc < 16; ++kc)
        aq[kc] = load_tile16x32(qp + kc * 32, C_DIM, lane);

    float mi[8], li[8];
    #pragma unroll
    for (int r = 0; r < 8; ++r) { mi[r] = -1e30f; li[r] = 0.f; }

    // 16 KB tile = 8 rounds x 128 threads x 16 B, fully coalesced.
    auto fill = [&](int buf, int kb) {
        const __bf16* src = kp + (size_t)kb * C_DIM;
        #pragma unroll
        for (int r = 0; r < 8; ++r) {
            int e = (r * 128 + tid) << 3;            // element index (8 bf16 / 16 B)
            async_copy_b128(sK_lds + buf * 16384u + (unsigned)(e << 1), src + e);
        }
    };
    auto s_tile = [&](const __bf16* kt) -> v8f {
        v8f s0 = {}, s1 = {}, s2 = {}, s3 = {};      // 4 independent chains
        #pragma unroll
        for (int kc = 0; kc < 16; kc += 4) {
            s0 = wmma_bf16(aq[kc + 0], load_tile16x32(kt + (kc + 0) * 32, C_DIM, lane), s0);
            s1 = wmma_bf16(aq[kc + 1], load_tile16x32(kt + (kc + 1) * 32, C_DIM, lane), s1);
            s2 = wmma_bf16(aq[kc + 2], load_tile16x32(kt + (kc + 2) * 32, C_DIM, lane), s2);
            s3 = wmma_bf16(aq[kc + 3], load_tile16x32(kt + (kc + 3) * 32, C_DIM, lane), s3);
        }
        return (s0 + s1) + (s2 + s3);
    };

    fill(0, 0);
    wait_async0();
    __syncthreads();

    for (int kb = 0; kb < HW_DIM; kb += 32) {
        fill(1, kb + 16);                            // odd tile -> buffer 1
        v8f sA = s_tile(&sK[0][0]);
        wait_async0();
        __syncthreads();

        if (kb + 32 < HW_DIM) fill(0, kb + 32);      // next even tile -> buffer 0
        v8f sB = s_tile(&sK[1][0]);

        #pragma unroll
        for (int r = 0; r < 8; ++r) {                // combined 32-key reduction
            float va = sA[r], vb = sB[r];
            float rm = fmaxf(va, vb);
            rm = fmaxf(rm, __shfl_xor(rm, 1, 32));
            rm = fmaxf(rm, __shfl_xor(rm, 2, 32));
            rm = fmaxf(rm, __shfl_xor(rm, 4, 32));
            rm = fmaxf(rm, __shfl_xor(rm, 8, 32));
            float mnew = fmaxf(mi[r], rm);
            float rs = __expf(va - mnew) + __expf(vb - mnew);
            rs += __shfl_xor(rs, 1, 32);
            rs += __shfl_xor(rs, 2, 32);
            rs += __shfl_xor(rs, 4, 32);
            rs += __shfl_xor(rs, 8, 32);
            li[r] = li[r] * __expf(mi[r] - mnew) + rs;
            mi[r] = mnew;
        }
        if (kb + 32 < HW_DIM) wait_async0();
        __syncthreads();
    }

    if ((lane & 15) == 0) {
        int hi = lane >> 4;
        #pragma unroll
        for (int r = 0; r < 8; ++r) {
            mstat[(size_t)t * HW_DIM + qbase + r + 8 * hi] = mi[r];
            lstat[(size_t)t * HW_DIM + qbase + r + 8 * hi] = li[r];
        }
    }
}

// ---------------------------------------------------------------------------
// Attention pass 2: O = P @ V for one 16-query block per workgroup.
// The Q tile is shared by all 4 waves -> staged ONCE into LDS via the async
// path (saves ~128 VGPRs/wave vs register-resident Q).  Per 128-key round
// every wave computes S/P for its own 32-key block, P goes through LDS
// (C-layout -> A-layout), then all waves WMMA their 128-ch O slice from V^T.
// grid (HW/16, T), block 128.
// ---------------------------------------------------------------------------
__global__ __launch_bounds__(128)
void attn_av_kernel(const __bf16* __restrict__ q, const __bf16* __restrict__ k,
                    const __bf16* __restrict__ vT, const float* __restrict__ mstat,
                    const float* __restrict__ lstat, __bf16* __restrict__ o) {
    __shared__ __bf16 sQ[16 * C_DIM];                // 16 KB, shared Q tile
    __shared__ __bf16 sP[4][16 * 32];                // 4 KB P tiles
    const int tid   = threadIdx.x;
    const int lane  = tid & 31;
    const int wave  = tid >> 5;
    const int t     = blockIdx.y;
    const int qbase = blockIdx.x * 16;
    const int hi    = lane >> 4;
    const int nn    = lane & 15;
    const __bf16* kp = k  + (size_t)t * HW_DIM * C_DIM;
    const __bf16* vp = vT + ((size_t)t * C_DIM + wave * 128) * HW_DIM;

    {   // async-stage the shared 16x512 Q tile (contiguous 16 KB)
        const __bf16* src = q + ((size_t)t * HW_DIM + qbase) * C_DIM;
        const unsigned sQ_lds = (unsigned)(uintptr_t)&sQ[0];
        #pragma unroll
        for (int r = 0; r < 8; ++r) {
            int e = (r * 128 + tid) << 3;
            async_copy_b128(sQ_lds + (unsigned)(e << 1), src + e);
        }
    }
    float mrow[8], rinv[8];
    #pragma unroll
    for (int r = 0; r < 8; ++r) {
        int row = qbase + r + 8 * hi;
        mrow[r] = mstat[(size_t)t * HW_DIM + row];
        rinv[r] = 1.0f / lstat[(size_t)t * HW_DIM + row];
    }
    wait_async0();
    __syncthreads();

    v8f oacc[8] = {};
    for (int kb0 = 0; kb0 < HW_DIM; kb0 += 128) {
        const int kb = kb0 + wave * 32;
        if (kb0 + 128 < HW_DIM)
            __builtin_prefetch(kp + (size_t)(kb + 128) * C_DIM, 0, 1);

        #pragma unroll
        for (int half = 0; half < 2; ++half) {
            const __bf16* kt = kp + (size_t)(kb + half * 16) * C_DIM;
            v8f s0 = {}, s1 = {};
            #pragma unroll
            for (int kc = 0; kc < 16; kc += 2) {
                s0 = wmma_bf16(load_tile16x32(sQ + (kc + 0) * 32, C_DIM, lane),
                               load_tile16x32(kt + (kc + 0) * 32, C_DIM, lane), s0);
                s1 = wmma_bf16(load_tile16x32(sQ + (kc + 1) * 32, C_DIM, lane),
                               load_tile16x32(kt + (kc + 1) * 32, C_DIM, lane), s1);
            }
            v8f s = s0 + s1;
            #pragma unroll
            for (int r = 0; r < 8; ++r) {
                float pv = __expf(s[r] - mrow[r]) * rinv[r];
                sP[wave][(r + 8 * hi) * 32 + half * 16 + nn] = (__bf16)pv;
            }
        }
        __syncthreads();

        #pragma unroll
        for (int j = 0; j < 4; ++j) {                // consume all 4 P tiles
            v16bf ap = load_tile16x32(&sP[j][0], 32, lane);
            const __bf16* vb = vp + kb0 + j * 32;
            #pragma unroll
            for (int nt = 0; nt < 8; ++nt)
                oacc[nt] = wmma_bf16(ap, load_tile16x32(vb + (size_t)nt * 16 * HW_DIM,
                                                        HW_DIM, lane), oacc[nt]);
        }
        __syncthreads();
    }

    #pragma unroll
    for (int nt = 0; nt < 8; ++nt) {
        int n = wave * 128 + nt * 16 + nn;
        #pragma unroll
        for (int r = 0; r < 8; ++r) {
            int m = qbase + r + 8 * hi;
            o[((size_t)t * HW_DIM + m) * C_DIM + n] = (__bf16)oacc[nt][r];
        }
    }
}

// ---------------------------------------------------------------------------
// Host-side orchestration.
// ---------------------------------------------------------------------------
extern "C" void kernel_launch(void* const* d_in, const int* in_sizes, int n_in,
                              void* d_out, int out_size, void* d_ws, size_t ws_size,
                              hipStream_t stream) {
    const float* x     = (const float*)d_in[0];
    const float* gamma = (const float*)d_in[1];
    const float* beta  = (const float*)d_in[2];
    const float* wq    = (const float*)d_in[3];
    const float* bq    = (const float*)d_in[4];
    const float* wk    = (const float*)d_in[5];
    const float* bk    = (const float*)d_in[6];
    const float* wv    = (const float*)d_in[7];
    const float* bv    = (const float*)d_in[8];
    const float* wp    = (const float*)d_in[9];
    const float* bp    = (const float*)d_in[10];
    float* out = (float*)d_out;

    char* ws = (char*)d_ws;
    size_t off = 0;
    auto alloc = [&](size_t bytes) -> void* {
        void* p = ws + off;
        off += (bytes + 255) & ~(size_t)255;
        return p;
    };
    const size_t ME = (size_t)M_DIM * C_DIM;
    __bf16* hn    = (__bf16*)alloc(ME * 2);
    __bf16* qb    = (__bf16*)alloc(ME * 2);
    __bf16* kb    = (__bf16*)alloc(ME * 2);
    __bf16* vTb   = (__bf16*)alloc(ME * 2);
    __bf16* ob    = (__bf16*)alloc(ME * 2);
    __bf16* wqT   = (__bf16*)alloc((size_t)C_DIM * C_DIM * 2);
    __bf16* wkT   = (__bf16*)alloc((size_t)C_DIM * C_DIM * 2);
    __bf16* wvT   = (__bf16*)alloc((size_t)C_DIM * C_DIM * 2);
    __bf16* wpT   = (__bf16*)alloc((size_t)C_DIM * C_DIM * 2);
    float*  psum  = (float*)alloc(N_GRP * 16 * 4);
    float*  psum2 = (float*)alloc(N_GRP * 16 * 4);
    float*  gmean = (float*)alloc(N_GRP * 4);
    float*  grstd = (float*)alloc(N_GRP * 4);
    float*  mstat = (float*)alloc((size_t)M_DIM * 4);
    float*  lstat = (float*)alloc((size_t)M_DIM * 4);
    (void)ws_size; (void)in_sizes; (void)n_in; (void)out_size;

    prep_w_kernel<<<(C_DIM * C_DIM) / 256, 256, 0, stream>>>(
        wq, wk, wv, wp, wqT, wkT, wvT, wpT);

    gn_partial_kernel<<<dim3(N_GRP, 16), 256, 0, stream>>>(x, psum, psum2);
    gn_finalize_kernel<<<1, N_GRP, 0, stream>>>(psum, psum2, gmean, grstd);
    gn_apply_kernel<<<dim3(HW_DIM / 32, C_DIM / 32, T_DIM), dim3(32, 8), 0, stream>>>(
        x, gamma, beta, gmean, grstd, hn);

    dim3 ggrid(C_DIM / 64, M_DIM / 128);
    gemm_kernel<3><<<ggrid, 256, 0, stream>>>(hn, wqT, bq, qb,  nullptr, nullptr); // q*scale
    gemm_kernel<0><<<ggrid, 256, 0, stream>>>(hn, wkT, bk, kb,  nullptr, nullptr);
    gemm_kernel<1><<<ggrid, 256, 0, stream>>>(hn, wvT, bv, vTb, nullptr, nullptr);

    attn_stats_kernel<<<dim3(HW_DIM / 64, T_DIM), 128, 0, stream>>>(qb, kb, mstat, lstat);
    attn_av_kernel<<<dim3(HW_DIM / 16, T_DIM), 128, 0, stream>>>(qb, kb, vTb,
                                                                 mstat, lstat, ob);

    gemm_kernel<2><<<ggrid, 256, 0, stream>>>(ob, wpT, bp, nullptr, out, x);
}
